// P2RSpatialLoss_32461362823617
// MI455X (gfx1250) — matrix operand order repaired and checked by reading, not compiled, not run on hardware
//
#include <hip/hip_runtime.h>
#include <hip/hip_bf16.h>

typedef __attribute__((ext_vector_type(16))) _Float16 v16h;
typedef __attribute__((ext_vector_type(8)))  _Float16 v8h;
typedef __attribute__((ext_vector_type(8)))  float    v8f;

#define B_   16
#define N_   1024
#define H_   512
#define W_   512
#define K_   N_
#define HW_  (H_ * W_)
#define SIGMA_ 8.0f

// ---------------------------------------------------------------------------
// 0. Init: zero output scalar and per-image max slots (bit pattern 0 == 0.0f,
//    valid because target/pred are non-negative).
// ---------------------------------------------------------------------------
__global__ void init_k(float* __restrict__ out, int* __restrict__ tmax,
                       int* __restrict__ pmax) {
  int t = threadIdx.x;
  if (t == 0) out[0] = 0.0f;
  if (t < B_) { tmax[t] = 0; pmax[t] = 0; }
}

// ---------------------------------------------------------------------------
// 1. Gaussian factor generation, stored K-contiguous in f16:
//    out[b][p][n] = exp(-(p - points[b][n][coord]/down)^2 * inv2s2)
//    (H_ == W_ == 512, so one kernel serves both gxT and gyT)
// ---------------------------------------------------------------------------
__global__ void __launch_bounds__(256)
gauss_k(const float* __restrict__ points, const int* __restrict__ down_div,
        const int* __restrict__ down_h, _Float16* __restrict__ out, int coord) {
  size_t idx = (size_t)blockIdx.x * blockDim.x + threadIdx.x; // b*P*N + p*N + n
  int    n   = (int)(idx & (N_ - 1));
  size_t pn  = idx >> 10;                 // N_ = 1024
  int    p   = (int)(pn & (H_ - 1));      // P  = 512
  int    b   = (int)(pn >> 9);

  float dh     = (float)down_h[0];
  float s      = SIGMA_ / dh;             // reference: s from down_h for BOTH axes
  float inv2s2 = 1.0f / (2.0f * s * s);
  float c      = points[((size_t)b * N_ + n) * 2 + coord] / (float)down_div[0];
  float d      = (float)p - c;
  out[idx]     = (_Float16)__expf(-d * d * inv2s2);
}

// ---------------------------------------------------------------------------
// 2. Per-image max of pred (non-negative -> integer-bit atomicMax).
// ---------------------------------------------------------------------------
__global__ void __launch_bounds__(256)
pmax_k(const float* __restrict__ pred, int* __restrict__ pmax) {
  size_t t  = (size_t)blockIdx.x * blockDim.x + threadIdx.x;
  size_t i4 = t * 4;
  int    b  = (int)(i4 >> 18);            // HW_ = 2^18
  float4 v  = *(const float4*)(pred + i4);
  float  m  = fmaxf(fmaxf(v.x, v.y), fmaxf(v.z, v.w));
  for (int off = 16; off > 0; off >>= 1)
    m = fmaxf(m, __shfl_xor(m, off, 32));
  if ((threadIdx.x & 31) == 0) atomicMax(pmax + b, __float_as_int(m));
}

// ---------------------------------------------------------------------------
// 3. Batched GEMM via WMMA with 2x2 register tiling (32x32 output per wave).
//    Both factors stored K-contiguous so every fragment load is a contiguous
//    16B/32B vector load ; A and B share one contiguous k order, k = 16*(lane/16) + s.
//    Each A fragment feeds 2 WMMAs and each B fragment feeds 2 WMMAs,
//    halving L2 panel traffic vs 1 tile/wave. 4 WMMAs per 32-deep K-step.
// ---------------------------------------------------------------------------
__device__ __forceinline__ v16h load_afrag(const _Float16* p) {
  // p pre-offset by k0 + koffA. Halfs 0..15 = K+0..15(+koff)
  v8h lo = *(const v8h*)(p);
  v8h hi = *(const v8h*)(p + 8);
  return __builtin_shufflevector(lo, hi, 0, 1, 2, 3, 4, 5, 6, 7,
                                         8, 9, 10, 11, 12, 13, 14, 15);
}

__global__ void __launch_bounds__(256)
gemm_k(const _Float16* __restrict__ gyT, const _Float16* __restrict__ gxT,
       float* __restrict__ target, int* __restrict__ tmax) {
  const int lane  = threadIdx.x & 31;
  const int wave  = threadIdx.x >> 5;
  const int gtile = blockIdx.x * 8 + wave;   // B_*16*16 = 4096 wave-tiles
  const int b     = gtile >> 8;              // 256 tiles per image
  const int tm    = (gtile >> 4) & 15;       // 32-row block
  const int tn    = gtile & 15;              // 32-col block

  const int mr    = lane & 15;
  const int koffA = (lane & 16) ? 16 : 0;    // A: lanes 0-15 K 0..15; 16-31 K 16..31
  const int koffB = (lane & 16) ? 16 : 0;    // B: lanes 0-15 K rows 0..15;  16-31 K rows 16..31

  const _Float16* __restrict__ Ap0 = gyT + ((size_t)b * H_ + tm * 32 + mr) * K_;
  const _Float16* __restrict__ Ap1 = Ap0 + (size_t)16 * K_;
  const _Float16* __restrict__ Bp0 = gxT + ((size_t)b * W_ + tn * 32 + mr) * K_;
  const _Float16* __restrict__ Bp1 = Bp0 + (size_t)16 * K_;

  v8f acc00 = {}, acc01 = {}, acc10 = {}, acc11 = {};
#pragma unroll 2
  for (int k0 = 0; k0 < K_; k0 += 32) {
    __builtin_prefetch(Ap0 + k0 + 512, 0, 3);  // near-cache (WGP) prefetch ahead
    __builtin_prefetch(Bp0 + k0 + 512, 0, 3);
    v16h a0 = load_afrag(Ap0 + k0 + koffA);
    v16h a1 = load_afrag(Ap1 + k0 + koffA);
    v16h b0 = *(const v16h*)(Bp0 + k0 + koffB);
    v16h b1 = *(const v16h*)(Bp1 + k0 + koffB);
    // 8 args: (neg_a, A, neg_b, B, c_mod, C, reuse_a, reuse_b)
    acc00 = __builtin_amdgcn_wmma_f32_16x16x32_f16(false, a0, false, b0, (short)0, acc00, false, false);
    acc01 = __builtin_amdgcn_wmma_f32_16x16x32_f16(false, a0, false, b1, (short)0, acc01, false, false);
    acc10 = __builtin_amdgcn_wmma_f32_16x16x32_f16(false, a1, false, b0, (short)0, acc10, false, false);
    acc11 = __builtin_amdgcn_wmma_f32_16x16x32_f16(false, a1, false, b1, (short)0, acc11, false, false);
  }

  // C/D layout: VGPR r -> lanes 0-15: M=r, N=lane; lanes 16-31: M=8+r, N=lane-16.
  const int col0 = tn * 32 + (lane & 15);
  const int row0 = tm * 32 + ((lane & 16) ? 8 : 0);
  float lmax = 0.0f;
#pragma unroll
  for (int r = 0; r < 8; ++r) {
    const size_t rb0 = ((size_t)b * H_ + row0 + r) * W_;
    const size_t rb1 = ((size_t)b * H_ + row0 + 16 + r) * W_;
    float v00 = acc00[r], v01 = acc01[r], v10 = acc10[r], v11 = acc11[r];
    target[rb0 + col0]      = v00;
    target[rb0 + col0 + 16] = v01;
    target[rb1 + col0]      = v10;
    target[rb1 + col0 + 16] = v11;
    lmax = fmaxf(lmax, fmaxf(fmaxf(v00, v01), fmaxf(v10, v11)));
  }
  for (int off = 16; off > 0; off >>= 1)
    lmax = fmaxf(lmax, __shfl_xor(lmax, off, 32));
  if (lane == 0) atomicMax(tmax + b, __float_as_int(lmax));
}

// ---------------------------------------------------------------------------
// 4. Normalized MSE reduction: sum((pred/pmax - tgt/tmax)^2) / (H*W*B)
// ---------------------------------------------------------------------------
__global__ void __launch_bounds__(256)
loss_k(const float* __restrict__ pred, const float* __restrict__ target,
       const int* __restrict__ tmax, const int* __restrict__ pmax,
       float* __restrict__ out) {
  __shared__ float wsum[8];
  size_t t  = (size_t)blockIdx.x * blockDim.x + threadIdx.x;
  size_t i4 = t * 4;
  int    b  = (int)(i4 >> 18);

  float tm  = __int_as_float(tmax[b]); tm = tm > 0.0f ? tm : 1.0f;
  float pm  = __int_as_float(pmax[b]); pm = pm > 0.0f ? pm : 1.0f;
  float rtm = 1.0f / tm, rpm = 1.0f / pm;

  float4 p  = *(const float4*)(pred + i4);
  float4 tg = *(const float4*)(target + i4);
  float d0 = p.x * rpm - tg.x * rtm;
  float d1 = p.y * rpm - tg.y * rtm;
  float d2 = p.z * rpm - tg.z * rtm;
  float d3 = p.w * rpm - tg.w * rtm;
  float s  = d0 * d0 + d1 * d1 + d2 * d2 + d3 * d3;

  for (int off = 16; off > 0; off >>= 1)
    s += __shfl_xor(s, off, 32);
  int lane = threadIdx.x & 31, wv = threadIdx.x >> 5;
  if (lane == 0) wsum[wv] = s;
  __syncthreads();
  if (threadIdx.x == 0) {
    float tot = 0.0f;
#pragma unroll
    for (int i = 0; i < 8; ++i) tot += wsum[i];
    atomicAdd(out, tot * (1.0f / ((float)HW_ * (float)B_)));
  }
}

// ---------------------------------------------------------------------------
// Launch. d_in: [0]=pred_density f32, [1]=points f32, [2]=down_h i32, [3]=down_w i32
// d_ws layout: gyT f16 (16MB) | gxT f16 (16MB) | target f32 (16MB) | tmax/pmax ints
// ---------------------------------------------------------------------------
extern "C" void kernel_launch(void* const* d_in, const int* in_sizes, int n_in,
                              void* d_out, int out_size, void* d_ws, size_t ws_size,
                              hipStream_t stream) {
  const float* pred   = (const float*)d_in[0];
  const float* points = (const float*)d_in[1];
  const int*   down_h = (const int*)d_in[2];
  const int*   down_w = (const int*)d_in[3];
  float*       out    = (float*)d_out;

  char*     ws     = (char*)d_ws;
  _Float16* gyT    = (_Float16*)ws;                                   // B*H*K f16
  _Float16* gxT    = (_Float16*)(ws + (size_t)16 * 1024 * 1024);      // B*W*K f16
  float*    target = (float*)   (ws + (size_t)32 * 1024 * 1024);      // B*H*W f32
  int*      tmax   = (int*)     (ws + (size_t)48 * 1024 * 1024);      // B ints
  int*      pmax   = tmax + B_;                                       // B ints

  init_k<<<1, 64, 0, stream>>>(out, tmax, pmax);

  const int genBlocks = (B_ * H_ * N_) / 256;                         // 32768
  gauss_k<<<genBlocks, 256, 0, stream>>>(points, down_w, down_h, gxT, 0);
  gauss_k<<<genBlocks, 256, 0, stream>>>(points, down_h, down_h, gyT, 1);

  pmax_k<<<(B_ * HW_) / (256 * 4), 256, 0, stream>>>(pred, pmax);     // 4096 blocks

  gemm_k<<<(B_ * 16 * 16) / 8, 256, 0, stream>>>(gyT, gxT, target, tmax); // 512 blocks

  loss_k<<<(B_ * HW_) / (256 * 4), 256, 0, stream>>>(pred, target, tmax, pmax, out);
}
